// VeRALayer_28690381537981
// MI455X (gfx1250) — compile-verified, matrix operand order checked by reading, not run on hardware
//
#include <hip/hip_runtime.h>

typedef __bf16 bf16_t;
typedef bf16_t v4bf  __attribute__((ext_vector_type(4)));
typedef bf16_t v8bf  __attribute__((ext_vector_type(8)));
typedef bf16_t v16bf __attribute__((ext_vector_type(16)));
typedef float  v8f   __attribute__((ext_vector_type(8)));

#define BM 64
#define BN 128
#define BK 64
#define LDSS 72  // bf16 row stride: rows 16B-aligned, 36-dword stride => conflict-free banks

// Split f32 -> (hi, lo) bf16 pair and store 4 elements to the hi/lo LDS tiles.
__device__ __forceinline__ void decomp_store(const float4 v,
                                             bf16_t* __restrict__ ph,
                                             bf16_t* __restrict__ pl) {
    v4bf h, l;
    h[0] = (bf16_t)v.x; h[1] = (bf16_t)v.y; h[2] = (bf16_t)v.z; h[3] = (bf16_t)v.w;
    l[0] = (bf16_t)(v.x - (float)h[0]);
    l[1] = (bf16_t)(v.y - (float)h[1]);
    l[2] = (bf16_t)(v.z - (float)h[2]);
    l[3] = (bf16_t)(v.w - (float)h[3]);
    *(v4bf*)ph = h;
    *(v4bf*)pl = l;
}

// A/B fragment per ISA 7.12.2: lane holds two contiguous 8-elem K runs
// at byte-aligned offsets base and base+16 elements -> two ds_load_b128.
__device__ __forceinline__ v16bf load_frag(const bf16_t* base) {
    v8bf lo = *(const v8bf*)(base);
    v8bf hi = *(const v8bf*)(base + 16);
    return __builtin_shufflevector(lo, hi, 0, 1, 2, 3, 4, 5, 6, 7,
                                           8, 9, 10, 11, 12, 13, 14, 15);
}

// C[M,N] = s[n]*scale * (X[M,K] @ W[N,K]^T), bf16x3 split-precision WMMA.
// Requires M%64==0, N%128==0, K%64==0 (holds: M=8192, N in {256,4096}, K in {4096,256}).
__global__ __launch_bounds__(256)
void gemm_xwt_colscale_bf16x3(const float* __restrict__ X,
                              const float* __restrict__ W,
                              const float* __restrict__ s,
                              float* __restrict__ C,
                              int M, int N, int K, float scale)
{
    __shared__ bf16_t Xh[BM * LDSS], Xl[BM * LDSS];
    __shared__ bf16_t Wh[BN * LDSS], Wl[BN * LDSS];

    const int t    = threadIdx.x;
    const int lane = t & 31;
    const int wave = t >> 5;

    const int m0 = blockIdx.y * BM;
    const int n0 = blockIdx.x * BN;

    const int tr = (wave & 3) * 16;    // wave's 16-row band (4 bands)
    const int ch = (wave >> 2) * 64;   // wave's 64-col band (4 x 16 tiles)

    const int lm = lane & 15;          // fragment row/col
    const int lh = lane >> 4;          // K-run selector (frags), M+8 (C/D)

    v8f acc[4] = {{}, {}, {}, {}};

    for (int k0 = 0; k0 < K; k0 += BK) {
        // ---- stage X tile: 64x64 f32 = 1024 float4, 4 per thread
#pragma unroll
        for (int i = 0; i < 4; ++i) {
            const int id = t + 256 * i;
            const int r = id >> 4, c = (id & 15) * 4;
            float4 v = *(const float4*)&X[(size_t)(m0 + r) * K + k0 + c];
            decomp_store(v, &Xh[r * LDSS + c], &Xl[r * LDSS + c]);
        }
        // ---- stage W tile: 128x64 f32 = 2048 float4, 8 per thread
#pragma unroll
        for (int i = 0; i < 8; ++i) {
            const int id = t + 256 * i;
            const int r = id >> 4, c = (id & 15) * 4;
            float4 v = *(const float4*)&W[(size_t)(n0 + r) * K + k0 + c];
            decomp_store(v, &Wh[r * LDSS + c], &Wl[r * LDSS + c]);
        }
        // ---- prefetch next K chunk toward near caches while this one computes
        if (k0 + BK < K) {
            __builtin_prefetch(&X[(size_t)(m0 + (t >> 4)) * K + k0 + BK + (t & 15) * 4], 0, 3);
            __builtin_prefetch(&W[(size_t)(n0 + (t >> 4)) * K + k0 + BK + (t & 15) * 4], 0, 3);
        }
        __syncthreads();

        // ---- two K=32 spans; 3 bf16 WMMAs per 16x16 tile per span (bf16x3)
#pragma unroll
        for (int sp = 0; sp < 2; ++sp) {
            const int kb = 32 * sp + 8 * lh;
            v16bf xh = load_frag(&Xh[(tr + lm) * LDSS + kb]);
            v16bf xl = load_frag(&Xl[(tr + lm) * LDSS + kb]);
#pragma unroll
            for (int tn = 0; tn < 4; ++tn) {
                const int wr = (ch + tn * 16 + lm) * LDSS + kb;
                v16bf wh = load_frag(&Wh[wr]);
                v16bf wl = load_frag(&Wl[wr]);
                acc[tn] = __builtin_amdgcn_wmma_f32_16x16x32_bf16(
                    false, xh, false, wh, (short)0, acc[tn], false, false);
                acc[tn] = __builtin_amdgcn_wmma_f32_16x16x32_bf16(
                    false, xh, false, wl, (short)0, acc[tn], false, false);
                acc[tn] = __builtin_amdgcn_wmma_f32_16x16x32_bf16(
                    false, xl, false, wh, (short)0, acc[tn], false, false);
            }
        }
        __syncthreads();
    }

    // ---- epilogue: per-column scale, direct f32 stores
#pragma unroll
    for (int tn = 0; tn < 4; ++tn) {
        const int col = n0 + ch + tn * 16 + lm;
        const float sc = s[col] * scale;
#pragma unroll
        for (int v = 0; v < 8; ++v) {
            C[(size_t)(m0 + tr + v + 8 * lh) * N + col] = acc[tn][v] * sc;
        }
    }
}

extern "C" void kernel_launch(void* const* d_in, const int* in_sizes, int n_in,
                              void* d_out, int out_size, void* d_ws, size_t ws_size,
                              hipStream_t stream) {
    (void)n_in; (void)out_size; (void)ws_size;

    const float* x   = (const float*)d_in[0];  // (4,2048,4096)
    const float* A   = (const float*)d_in[1];  // (256,4096)  row-major (rank,in)
    const float* B   = (const float*)d_in[2];  // (4096,256)  row-major (out,rank)
    const float* d_A = (const float*)d_in[3];  // (256,)
    const float* d_B = (const float*)d_in[4];  // (4096,)
    float* out = (float*)d_out;                // (4,2048,4096)
    float* T   = (float*)d_ws;                 // (8192,256) intermediate, 8 MB

    const int K1 = 4096;                       // input dim
    const int R  = 256;                        // rank
    const int N2 = 4096;                       // output dim
    const int M  = in_sizes[0] / K1;           // 8192 tokens
    const float SCALE = 32.0f / 256.0f;

    dim3 blk(256);

    // Stage 1: T = (x @ A^T) * (d_A * SCALE) per column     M x R, K = 4096
    dim3 g1(R / BN, M / BM);
    gemm_xwt_colscale_bf16x3<<<g1, blk, 0, stream>>>(x, A, d_A, T, M, R, K1, SCALE);

    // Stage 2: out = (T @ B^T) * d_B per column             M x N2, K = 256
    dim3 g2(N2 / BN, M / BM);
    gemm_xwt_colscale_bf16x3<<<g2, blk, 0, stream>>>(T, B, d_B, out, M, N2, R, 1.0f);
}
